// MoELayer_12824772346577
// MI455X (gfx1250) — matrix operand order repaired and checked by reading, not compiled, not run on hardware
//
#include <hip/hip_runtime.h>

// ---------------- problem constants ----------------
#define DIMX 1024
#define HIDX 2048
#define NEXP 8
#define NTOK 8192            // 4 * 2048 tokens
#define BM   32              // tokens per FFN tile (2 WMMA row-blocks)
#define NTILES (NTOK / BM)   // 256 worst-case tiles per expert
#define NCHUNK (HIDX / 64)   // 32 hidden chunks of 64

// ---------------- LDS layout (bytes, every region 16B aligned) ----------------
#define XS_ROW    1032                     // bf16 elems per X row (2064B row: /16 ok, 4-bank skew)
#define XS_BYTES  (BM * XS_ROW * 2)        // 66048
#define WT_ST     136                      // stage-1 weight tile k-stride (272B row)
#define W3_ST     72                       // stage-2 w3 tile k-stride (144B row)
#define W3_COLS   256                      // stage-2 staging quarter width
#define WU_BYTES  (W3_COLS * W3_ST * 2)    // 36864 (covers stage1: 2*64*136*2 = 34816)
#define A12_BYTES (2 * BM * 64 * 4)        // 16384
#define HS_ST     72
#define HS_BYTES  (BM * HS_ST * 2)         // 4608
#define B3_BYTES  (DIMX * 4)               // 4096
#define CW_BYTES  (BM * 4)                 // 128
#define TOK_BYTES (BM * 4)                 // 128
#define SMEM_TOTAL (XS_BYTES + WU_BYTES + A12_BYTES + HS_BYTES + B3_BYTES + CW_BYTES + TOK_BYTES)

typedef __attribute__((ext_vector_type(16))) __bf16 v16bf;
typedef __attribute__((ext_vector_type(8)))  __bf16 v8bf;
typedef __attribute__((ext_vector_type(4)))  __bf16 v4bf;
typedef __attribute__((ext_vector_type(2)))  __bf16 v2bf;
typedef __attribute__((ext_vector_type(8)))  float  v8f;
typedef __attribute__((ext_vector_type(4)))  float  v4f;

__device__ __forceinline__ v16bf ld_frag(const __bf16* p0, const __bf16* p1) {
  v8bf a = *(const v8bf*)p0;
  v8bf b = *(const v8bf*)p1;
  return __builtin_shufflevector(a, b, 0,1,2,3,4,5,6,7,8,9,10,11,12,13,14,15);
}

// ---------------- kernel 0: zero output + counters ----------------
__global__ void moe_init_kernel(float* __restrict__ out, int* __restrict__ cnt,
                                float* __restrict__ usage) {
  const size_t stride = (size_t)gridDim.x * blockDim.x;
  for (size_t i = (size_t)blockIdx.x * blockDim.x + threadIdx.x;
       i < (size_t)NTOK * DIMX; i += stride)
    out[i] = 0.0f;
  if (blockIdx.x == 0 && threadIdx.x < NEXP) {
    cnt[threadIdx.x] = 0;
    usage[threadIdx.x] = 0.0f;
  }
}

// ---------------- kernel 1: gating (softmax + top-2 dispatch) ----------------
__global__ __launch_bounds__(256)
void moe_gate_kernel(const float* __restrict__ x, const float* __restrict__ gw,
                     const float* __restrict__ gb, float* __restrict__ combine,
                     int* __restrict__ cnt, float* __restrict__ usage,
                     int* __restrict__ lists) {
  __shared__ float GW[DIMX * NEXP];          // 32KB
  const int tid = threadIdx.x;
  for (int i = tid; i < DIMX * NEXP; i += 256) GW[i] = gw[i];
  __syncthreads();

  const int lane = tid & 31;
  const int wv   = tid >> 5;
  const int t    = blockIdx.x * 8 + wv;       // one wave32 per token
  const float* xr = x + (size_t)t * DIMX;

  float acc[NEXP] = {};
  for (int k = lane; k < DIMX; k += 32) {
    const float xv = xr[k];
    #pragma unroll
    for (int e = 0; e < NEXP; ++e) acc[e] += xv * GW[k * NEXP + e];
  }
  #pragma unroll
  for (int e = 0; e < NEXP; ++e)
    #pragma unroll
    for (int off = 16; off > 0; off >>= 1)
      acc[e] += __shfl_xor(acc[e], off, 32);

  if (lane == 0) {
    float mx = -1e30f;
    #pragma unroll
    for (int e = 0; e < NEXP; ++e) { acc[e] += gb[e]; mx = fmaxf(mx, acc[e]); }
    float den = 0.0f, sc[NEXP];
    #pragma unroll
    for (int e = 0; e < NEXP; ++e) { sc[e] = __expf(acc[e] - mx); den += sc[e]; }
    const float inv = 1.0f / den;
    int i1 = 0; float s1 = -1.0f;
    #pragma unroll
    for (int e = 0; e < NEXP; ++e) {
      sc[e] *= inv;
      atomicAdd(&usage[e], sc[e]);
      if (sc[e] > s1) { s1 = sc[e]; i1 = e; }
    }
    int i2 = -1; float s2 = -1.0f;
    #pragma unroll
    for (int e = 0; e < NEXP; ++e)
      if (e != i1 && sc[e] > s2) { s2 = sc[e]; i2 = e; }
    #pragma unroll
    for (int e = 0; e < NEXP; ++e)
      combine[(size_t)t * NEXP + e] = (e == i1) ? s1 : ((e == i2) ? s2 : 0.0f);
    int p1 = atomicAdd(&cnt[i1], 1); lists[i1 * NTOK + p1] = t;
    int p2 = atomicAdd(&cnt[i2], 1); lists[i2 * NTOK + p2] = t;
  }
}

// ---------------- kernel 2: load-balance loss ----------------
__global__ void moe_loss_kernel(const float* __restrict__ usage, float* __restrict__ loss) {
  if (blockIdx.x == 0 && threadIdx.x == 0) {
    float l = 0.0f;
    for (int e = 0; e < NEXP; ++e) {
      const float u = usage[e] / (float)NTOK;
      l += u * __logf(u + 1e-9f);
    }
    *loss = l;
  }
}

// ---------------- kernel 3: grouped fused SwiGLU FFN (bf16 WMMA, f32 accum) ----------------
__global__ __launch_bounds__(256, 1)
void moe_ffn_kernel(const float* __restrict__ x,
                    const float* __restrict__ w1, const float* __restrict__ b1,
                    const float* __restrict__ w2, const float* __restrict__ b2,
                    const float* __restrict__ w3, const float* __restrict__ b3,
                    const float* __restrict__ combine,
                    const int* __restrict__ cnt, const int* __restrict__ lists,
                    float* __restrict__ out) {
  extern __shared__ char smem[];
  __bf16* Xs  = (__bf16*)smem;
  __bf16* WU  = (__bf16*)(smem + XS_BYTES);   // stage1 Wt (2x64x128) / stage2 W3t (256x64)
  float*  A12 = (float*)(smem + XS_BYTES + WU_BYTES);
  __bf16* Hs  = (__bf16*)(smem + XS_BYTES + WU_BYTES + A12_BYTES);
  float*  B3s = (float*)(smem + XS_BYTES + WU_BYTES + A12_BYTES + HS_BYTES);
  float*  Cw  = (float*)(smem + XS_BYTES + WU_BYTES + A12_BYTES + HS_BYTES + B3_BYTES);
  int*    Tok = (int*)(smem + XS_BYTES + WU_BYTES + A12_BYTES + HS_BYTES + B3_BYTES + CW_BYTES);

  const int e    = blockIdx.x >> 8;          // / NTILES
  const int tile = blockIdx.x & (NTILES - 1);
  const int ne   = cnt[e];
  if (tile * BM >= ne) return;               // empty tile for this expert

  const int tid   = threadIdx.x;
  const int lane  = tid & 31;
  const int wv    = tid >> 5;                // 8 waves
  const int lhalf = (lane < 16) ? 0 : 1;
  const int l15   = lane & 15;

  if (tid < BM) {
    const int idx = tile * BM + tid;
    const int tok = lists[e * NTOK + ((idx < ne) ? idx : tile * BM)];
    Tok[tid] = tok;
    Cw[tid]  = (idx < ne) ? combine[(size_t)tok * NEXP + e] : 0.0f;  // masked rows contribute 0
  }
  __syncthreads();
  for (int i = tid; i < DIMX; i += 256) B3s[i] = b3[(size_t)e * DIMX + i];
  for (int i = tid; i < BM * (DIMX / 4); i += 256) {   // vectorized X gather -> bf16
    const int m = i >> 8, k4 = (i & 255) * 4;
    const v4f v = *(const v4f*)&x[(size_t)Tok[m] * DIMX + k4];
    v4bf t; t[0] = (__bf16)v[0]; t[1] = (__bf16)v[1]; t[2] = (__bf16)v[2]; t[3] = (__bf16)v[3];
    *(v4bf*)&Xs[m * XS_ROW + k4] = t;
  }
  __syncthreads();

  const float* w1e = w1 + (size_t)e * DIMX * HIDX;
  const float* w2e = w2 + (size_t)e * DIMX * HIDX;
  const float* w3e = w3 + (size_t)e * HIDX * DIMX;
  const int mat = wv & 1;                    // 0 -> w1 path, 1 -> w2 (gate) path
  const int nt1 = wv >> 1;                   // stage-1 n-tile (0..3)

  const v8f z8 = {0.f,0.f,0.f,0.f,0.f,0.f,0.f,0.f};
  v8f Y[16] = {z8,z8,z8,z8,z8,z8,z8,z8,z8,z8,z8,z8,z8,z8,z8,z8}; // 2 x (16x128) f32 per wave

  #pragma unroll 1
  for (int hc = 0; hc < NCHUNK; ++hc) {
    const int hb = hc * 64;

    // ---- stage 1: (A1|A2) = X(32x1024) @ w{1,2}[:, hb:hb+64] ----
    v8f accA = z8, accB = z8;                // two M-subtiles
    #pragma unroll 1
    for (int slab = 0; slab < DIMX / 128; ++slab) {
      __syncthreads();                       // WU reuse fence
      const int kg = slab * 128;
      // 4x4-block transposed staging: coalesced b128 loads, packed cvt, b64 stores
      #pragma unroll
      for (int it = 0; it < 4; ++it) {
        const int i  = tid + it * 256;       // 1024 blocks of 4d x 4h
        const int m2 = i >> 9;
        const int r  = i & 511;
        const int d0 = (r >> 4) * 4;
        const int h0 = (r & 15) * 4;
        const float* wp = m2 ? w2e : w1e;
        v4f vr[4];
        #pragma unroll
        for (int dd = 0; dd < 4; ++dd) {
          const size_t go = (size_t)(kg + d0 + dd) * HIDX + hb + h0;
          vr[dd] = *(const v4f*)&wp[go];
          __builtin_prefetch(&wp[go + (size_t)128 * HIDX], 0, 1);   // speculative, next slab
        }
        #pragma unroll
        for (int hh = 0; hh < 4; ++hh) {
          v4bf t;
          t[0] = (__bf16)vr[0][hh]; t[1] = (__bf16)vr[1][hh];
          t[2] = (__bf16)vr[2][hh]; t[3] = (__bf16)vr[3][hh];
          *(v4bf*)&WU[(m2 * 64 + h0 + hh) * WT_ST + d0] = t;
        }
      }
      __syncthreads();
      #pragma unroll
      for (int kk = 0; kk < 4; ++kk) {
        const int ka = kg + kk * 32 + lhalf * 8;            // A: K 0-7/16-23 vs 8-15/24-31
        const v16bf A0 = ld_frag(&Xs[l15 * XS_ROW + ka], &Xs[l15 * XS_ROW + ka + 16]);
        const v16bf A1 = ld_frag(&Xs[(l15 + 16) * XS_ROW + ka], &Xs[(l15 + 16) * XS_ROW + ka + 16]);
        const int kb = kk * 32 + lhalf * 8;                 // B: K 0-7/16-23 vs 8-15/24-31
        const __bf16* bp = &WU[(mat * 64 + nt1 * 16 + l15) * WT_ST + kb];
        const v16bf B = ld_frag(bp, bp + 16);               // shared across both M-subtiles
        accA = __builtin_amdgcn_wmma_f32_16x16x32_bf16(false, A0, false, B, (short)0, accA, false, false);
        accB = __builtin_amdgcn_wmma_f32_16x16x32_bf16(false, A1, false, B, (short)0, accB, false, false);
      }
    }
    {   // spill the two f32 stage-1 tiles (C layout: vgpr r + lane half -> row)
      const int mrow = lhalf * 8;
      #pragma unroll
      for (int r = 0; r < 8; ++r) {
        A12[(mat * BM + mrow + r) * 64 + nt1 * 16 + l15]      = accA[r];
        A12[(mat * BM + 16 + mrow + r) * 64 + nt1 * 16 + l15] = accB[r];
      }
    }
    __syncthreads();

    // ---- SwiGLU elementwise: h = (X@w1 + b1) * sigmoid(X@w2 + b2) ----
    #pragma unroll
    for (int it = 0; it < (BM * 32) / 256; ++it) {          // pairs of h
      const int i = tid + it * 256;
      const int m = i >> 5, h0 = (i & 31) * 2;
      v2bf t;
      #pragma unroll
      for (int hh = 0; hh < 2; ++hh) {
        const float s1 = A12[(0 * BM + m) * 64 + h0 + hh] + b1[(size_t)e * HIDX + hb + h0 + hh];
        const float s2 = A12[(1 * BM + m) * 64 + h0 + hh] + b2[(size_t)e * HIDX + hb + h0 + hh];
        const float g  = 1.0f / (1.0f + __expf(-s2));
        t[hh] = (__bf16)(s1 * g);
      }
      *(v2bf*)&Hs[m * HS_ST + h0] = t;
    }

    // ---- stage 2: Y += h(32x64) @ w3[hb:hb+64, :] in 4 quarters of 256 cols ----
    #pragma unroll
    for (int nh = 0; nh < 4; ++nh) {
      __syncthreads();                       // Hs visible + WU reuse fence
      const int nbase = nh * W3_COLS;
      #pragma unroll
      for (int it = 0; it < 4; ++it) {       // 1024 blocks of 4k x 4n
        const int i  = tid + it * 256;
        const int k0 = (i >> 6) * 4;
        const int n0 = (i & 63) * 4;
        v4f vr[4];
        #pragma unroll
        for (int kq = 0; kq < 4; ++kq)
          vr[kq] = *(const v4f*)&w3e[(size_t)(hb + k0 + kq) * DIMX + nbase + n0];
        #pragma unroll
        for (int nn = 0; nn < 4; ++nn) {
          v4bf t;
          t[0] = (__bf16)vr[0][nn]; t[1] = (__bf16)vr[1][nn];
          t[2] = (__bf16)vr[2][nn]; t[3] = (__bf16)vr[3][nn];
          *(v4bf*)&WU[(n0 + nn) * W3_ST + k0] = t;
        }
      }
      __syncthreads();
      #pragma unroll
      for (int j = 0; j < 2; ++j) {
        const int ncolB = (wv + 8 * j) * 16 + l15;          // col within quarter
        #pragma unroll
        for (int kk = 0; kk < 2; ++kk) {
          const int ka = kk * 32 + lhalf * 8;
          const v16bf A0 = ld_frag(&Hs[l15 * HS_ST + ka], &Hs[l15 * HS_ST + ka + 16]);
          const v16bf A1 = ld_frag(&Hs[(l15 + 16) * HS_ST + ka], &Hs[(l15 + 16) * HS_ST + ka + 16]);
          const int kb = kk * 32 + lhalf * 8;
          const __bf16* bp = &WU[ncolB * W3_ST + kb];
          const v16bf B = ld_frag(bp, bp + 16);             // shared across both M-subtiles
          Y[0 * 8 + nh * 2 + j] = __builtin_amdgcn_wmma_f32_16x16x32_bf16(
              false, A0, false, B, (short)0, Y[0 * 8 + nh * 2 + j], false, false);
          Y[1 * 8 + nh * 2 + j] = __builtin_amdgcn_wmma_f32_16x16x32_bf16(
              false, A1, false, B, (short)0, Y[1 * 8 + nh * 2 + j], false, false);
        }
      }
    }
  }

  // ---- epilogue: out[tok] += combine * (Y + b3) ----
  const int mrow = lhalf * 8;
  #pragma unroll
  for (int ms = 0; ms < 2; ++ms) {
    #pragma unroll
    for (int q = 0; q < 8; ++q) {            // q = nh*2 + j  ->  gtile = (q>>1)*16 + wv + 8*(q&1)
      const int gt = (q >> 1) * 16 + wv + 8 * (q & 1);
      const int n  = gt * 16 + l15;
      const float bb = B3s[n];
      #pragma unroll
      for (int r = 0; r < 8; ++r) {
        const int m = ms * 16 + mrow + r;
        atomicAdd(&out[(size_t)Tok[m] * DIMX + n], (Y[ms * 8 + q][r] + bb) * Cw[m]);
      }
    }
  }
}

// ---------------- host launcher ----------------
extern "C" void kernel_launch(void* const* d_in, const int* in_sizes, int n_in,
                              void* d_out, int out_size, void* d_ws, size_t ws_size,
                              hipStream_t stream) {
  const float* x  = (const float*)d_in[0];
  const float* gw = (const float*)d_in[1];
  const float* gb = (const float*)d_in[2];
  const float* w1 = (const float*)d_in[3];
  const float* b1 = (const float*)d_in[4];
  const float* w2 = (const float*)d_in[5];
  const float* b2 = (const float*)d_in[6];
  const float* w3 = (const float*)d_in[7];
  const float* b3 = (const float*)d_in[8];
  float* out = (float*)d_out;

  // workspace layout (~512KB): combine | cnt | usage | lists
  float* combine = (float*)d_ws;
  int*   cnt     = (int*)(combine + (size_t)NTOK * NEXP);
  float* usage   = (float*)(cnt + NEXP);
  int*   lists   = (int*)(usage + NEXP);

  (void)hipFuncSetAttribute((const void*)moe_ffn_kernel,
                            hipFuncAttributeMaxDynamicSharedMemorySize, SMEM_TOTAL);

  moe_init_kernel<<<1024, 256, 0, stream>>>(out, cnt, usage);
  moe_gate_kernel<<<NTOK / 8, 256, 0, stream>>>(x, gw, gb, combine, cnt, usage, lists);
  moe_loss_kernel<<<1, 32, 0, stream>>>(usage, out + (size_t)NTOK * DIMX);
  moe_ffn_kernel<<<NEXP * NTILES, 256, SMEM_TOTAL, stream>>>(
      x, w1, b1, w2, b2, w3, b3, combine, cnt, lists, out);
}
